// MultiheadAttention_25984552140973
// MI455X (gfx1250) — matrix operand order repaired and checked by reading, not compiled, not run on hardware
//
#include <hip/hip_runtime.h>
#include <stdint.h>

typedef __attribute__((ext_vector_type(16))) _Float16 v16h;
typedef __attribute__((ext_vector_type(8)))  float    v8f;

#define N_HEADS 8
#define EMBED   64
#define BATCH   2
#define SEQ     2048
#define FEAT    512
#define QKV_COLS (EMBED * N_HEADS * 3)   // 1536

static __device__ __forceinline__ v8f wmma_f16(v16h a, v16h b, v8f c) {
    return __builtin_amdgcn_wmma_f32_16x16x32_f16(
        /*neg_a=*/false, a, /*neg_b=*/false, b,
        /*c_mod=*/(short)0, c, /*reuse_a=*/false, /*reuse_b=*/false);
}

// LDS byte offset of a generic pointer to __shared__ (addrspacecast 3->0 puts
// the LDS offset in the low 32 bits of the flat address).
static __device__ __forceinline__ uint32_t lds_off(const void* p) {
    return (uint32_t)(uintptr_t)p;
}

// GLOBAL_LOAD_ASYNC_TO_LDS_B128 (GV mode, saddr=off): vdst = per-lane LDS byte
// address, vaddr = per-lane 64-bit global address.  Tracked with ASYNCcnt.
static __device__ __forceinline__ void async_copy16(uint32_t lds, uint64_t ga) {
    asm volatile("global_load_async_to_lds_b128 %0, %1, off"
                 :: "v"(lds), "v"(ga) : "memory");
}
static __device__ __forceinline__ void wait_async(void) {
    asm volatile("s_wait_asynccnt 0" ::: "memory");
}
static __device__ __forceinline__ void wait_async_keep4(void) {
    asm volatile("s_wait_asynccnt 4" ::: "memory");   // retire oldest tile only
}

// Load 16 contiguous halfs into a fragment register pair set.
static __device__ __forceinline__ v16h load_frag16(const _Float16* p) {
    v16h f;
    #pragma unroll
    for (int v = 0; v < 16; ++v) f[v] = p[v];
    return f;
}

// ---------------------------------------------------------------------------
// Kernel 1: QKV projection.  emb = x @ W + b, scattered into q/k [B,H,S,E]
// and v [B,H,E,S] (transposed, so attention can async-copy V tiles straight).
// M = B*S = 4096, N = 1536, K = 512.  WG: 64x64, 4 waves x (16M x 64N).
// ---------------------------------------------------------------------------
__global__ __launch_bounds__(128) void qkv_gemm_kernel(
        const float* __restrict__ x, const float* __restrict__ W,
        const float* __restrict__ bias,
        _Float16* __restrict__ qh, _Float16* __restrict__ kh,
        _Float16* __restrict__ vh) {
    __shared__ _Float16 As[64 * 32];    // [m][k]
    __shared__ _Float16 Bst[64 * 32];   // [n][k]  (transposed at staging)

    const int tid  = threadIdx.x;
    const int lane = tid & 31;
    const int wv   = tid >> 5;
    const int nn   = lane & 15;
    const int hf   = lane >> 4;
    const int mbase = blockIdx.y * 64;
    const int nbase = blockIdx.x * 64;

    v8f acc[4] = {};

    for (int k0 = 0; k0 < FEAT; k0 += 32) {
        // Stage A tile [64m x 32k], f32 -> f16, contiguous LDS rows.
        {
            const int r = tid >> 1, c0 = (tid & 1) * 16;
            const float* src = x + (size_t)(mbase + r) * FEAT + k0 + c0;
            #pragma unroll
            for (int i = 0; i < 16; ++i) As[r * 32 + c0 + i] = (_Float16)src[i];
        }
        // Stage B tile transposed: read W row (k = r), write column-strided.
        {
            const int r = tid >> 2, c0 = (tid & 3) * 16;
            const float* src = W + (size_t)(k0 + r) * QKV_COLS + nbase + c0;
            #pragma unroll
            for (int i = 0; i < 16; ++i) Bst[(c0 + i) * 32 + r] = (_Float16)src[i];
        }
        __syncthreads();

        // Preload A fragment + all four B fragments (one DS clause), then
        // issue the four WMMAs back-to-back (D->C chaining needs no NOPs).
        v16h af;
        {
            const _Float16* arow = &As[(wv * 16 + nn) * 32];
            #pragma unroll
            for (int v = 0; v < 8; ++v) {
                const int k = (hf << 4) + (v << 1);
                af[2 * v]     = arow[k];
                af[2 * v + 1] = arow[k + 1];
            }
        }
        v16h bf[4];
        #pragma unroll
        for (int nt = 0; nt < 4; ++nt)
            bf[nt] = load_frag16(&Bst[(nt * 16 + nn) * 32 + (hf << 4)]);
        #pragma unroll
        for (int nt = 0; nt < 4; ++nt)
            acc[nt] = wmma_f16(af, bf[nt], acc[nt]);
        __syncthreads();
    }

    // Epilogue: +bias, f32->f16, branchless scatter.  f = e*24 + h*3 + c.
    const int bb    = mbase >> 11;
    const int sbase = (mbase & (SEQ - 1)) + wv * 16 + (hf << 3);
    #pragma unroll
    for (int nt = 0; nt < 4; ++nt) {
        const int gn  = nbase + nt * 16 + nn;
        const int e   = gn / 24;
        const int rem = gn - e * 24;
        const int h   = rem / 3;
        const int c   = rem - h * 3;
        const size_t hd  = (size_t)(bb * N_HEADS + h) * (SEQ * EMBED);
        const size_t c0  = hd + ((c < 2) ? (size_t)e : (size_t)e * SEQ);
        const size_t str = (c < 2) ? (size_t)EMBED : (size_t)1;
        _Float16* p = (c == 0) ? qh : ((c == 1) ? kh : vh);
        const float bval = bias[gn];
        #pragma unroll
        for (int v = 0; v < 8; ++v)
            p[c0 + (size_t)(sbase + v) * str] = (_Float16)(acc[nt][v] + bval);
    }
}

// ---------------------------------------------------------------------------
// Kernel 2: flash attention.  One WG per (b, h, 64 query rows); 4 waves x 16
// rows; KV tiles of 32 double-buffered in LDS via async global->LDS copies.
// ---------------------------------------------------------------------------
__global__ __launch_bounds__(128) void attn_kernel(
        const _Float16* __restrict__ qh, const _Float16* __restrict__ kh,
        const _Float16* __restrict__ vh, float* __restrict__ out) {
    __shared__ _Float16 Ks[2][32 * 64];     // [buf][kv][e]
    __shared__ _Float16 Vt[2][64 * 32];     // [buf][e][kv]
    __shared__ _Float16 Ps[4][16 * 32];     // per-wave P staging [m][kv]

    const int tid  = threadIdx.x;
    const int lane = tid & 31;
    const int wv   = tid >> 5;
    const int nn   = lane & 15;
    const int hf   = lane >> 4;

    const int blk = blockIdx.x;
    const int qt  = blk & 31;
    const int bh  = blk >> 5;
    const int b   = bh >> 3;
    const int h   = bh & 7;
    const int qbase = qt * 64 + wv * 16;
    const size_t head_off = (size_t)(b * N_HEADS + h) * SEQ * EMBED;

    const uint64_t k_ga = (uint64_t)(uintptr_t)(kh + head_off);
    const uint64_t v_ga = (uint64_t)(uintptr_t)(vh + head_off);

    // Issue one KV tile (4 async b128 per thread: 2 K units + 2 V units).
    auto issue_tile = [&](int kv0, int buf) {
        const uint32_t kb = lds_off(Ks[buf]);
        const uint32_t vb = lds_off(Vt[buf]);
        const uint32_t u0 = (uint32_t)tid, u1 = (uint32_t)tid + 128;
        async_copy16(kb + u0 * 16, k_ga + (uint64_t)kv0 * 128 + u0 * 16);
        async_copy16(kb + u1 * 16, k_ga + (uint64_t)kv0 * 128 + u1 * 16);
        #pragma unroll
        for (int j = 0; j < 2; ++j) {
            const uint32_t u = (uint32_t)tid + j * 128;
            const uint32_t e = u >> 2, cu = u & 3;
            async_copy16(vb + u * 16,
                         v_ga + (uint64_t)e * (SEQ * 2) +
                             (uint64_t)kv0 * 2 + cu * 16);
        }
    };

    // Q A-fragments (global, contiguous 16B chunks per lane).
    v16h aq[2];
    {
        const _Float16* qrow = qh + head_off + (size_t)(qbase + nn) * EMBED;
        #pragma unroll
        for (int ks = 0; ks < 2; ++ks) {
            #pragma unroll
            for (int v = 0; v < 8; ++v) {
                const int k = ks * 32 + (hf << 4) + (v << 1);
                aq[ks][2 * v]     = qrow[k];
                aq[ks][2 * v + 1] = qrow[k + 1];
            }
        }
    }

    float Mrow[8], Lrow[8];
    v8f O[4] = {};
    #pragma unroll
    for (int v = 0; v < 8; ++v) { Mrow[v] = -1e30f; Lrow[v] = 0.0f; }

    issue_tile(0, 0);

    const int NT = SEQ / 32;
    for (int it = 0; it < NT; ++it) {
        const int buf = it & 1;
        if (it + 1 < NT) {
            issue_tile((it + 1) * 32, buf ^ 1);   // prefetch next tile
            wait_async_keep4();                   // retire current tile only
        } else {
            wait_async();
        }
        __syncthreads();                          // tile visible to all waves

        // S = Q K^T: preload all four K B-fragments, then 4 WMMAs.
        v16h bk[4];
        #pragma unroll
        for (int nt = 0; nt < 2; ++nt)
            #pragma unroll
            for (int ks = 0; ks < 2; ++ks)
                bk[nt * 2 + ks] = load_frag16(
                    &Ks[buf][(nt * 16 + nn) * 64 + ks * 32 + (hf << 4)]);
        v8f S[2] = {};
        #pragma unroll
        for (int nt = 0; nt < 2; ++nt)
            #pragma unroll
            for (int ks = 0; ks < 2; ++ks)
                S[nt] = wmma_f16(aq[ks], bk[nt * 2 + ks], S[nt]);

        // Online softmax: row m = v + 8*half lives in slot v; reduce over the
        // 16 lanes holding its columns.
        float p0[8], p1[8], sc[8];
        #pragma unroll
        for (int v = 0; v < 8; ++v) {
            float t = fmaxf(S[0][v], S[1][v]);
            #pragma unroll
            for (int off = 8; off >= 1; off >>= 1)
                t = fmaxf(t, __shfl_xor(t, off, 32));
            const float mn = fmaxf(Mrow[v], t);
            sc[v] = __expf(Mrow[v] - mn);
            p0[v] = __expf(S[0][v] - mn);
            p1[v] = __expf(S[1][v] - mn);
            float r = p0[v] + p1[v];
            #pragma unroll
            for (int off = 8; off >= 1; off >>= 1)
                r += __shfl_xor(r, off, 32);
            Lrow[v] = Lrow[v] * sc[v] + r;
            Mrow[v] = mn;
        }
        #pragma unroll
        for (int et = 0; et < 4; ++et)
            #pragma unroll
            for (int v = 0; v < 8; ++v) O[et][v] *= sc[v];

        // Stage P (D layout) -> per-wave LDS patch -> reload in A layout.
        // DS ops of one wave are processed in order; compiler fence only.
        _Float16* Pw = Ps[wv];
        #pragma unroll
        for (int v = 0; v < 8; ++v) {
            const int m = v + (hf << 3);
            Pw[m * 32 + nn]      = (_Float16)p0[v];
            Pw[m * 32 + 16 + nn] = (_Float16)p1[v];
        }
        asm volatile("" ::: "memory");

        v16h pa;
        {
            const _Float16* prow = &Pw[nn * 32];
            #pragma unroll
            for (int v = 0; v < 8; ++v) {
                const int k = (hf << 4) + (v << 1);
                pa[2 * v]     = prow[k];
                pa[2 * v + 1] = prow[k + 1];
            }
        }
        // O += P V: preload all four V B-fragments, then 4 WMMAs.
        v16h bv[4];
        #pragma unroll
        for (int et = 0; et < 4; ++et)
            bv[et] = load_frag16(&Vt[buf][(et * 16 + nn) * 32 + (hf << 4)]);
        #pragma unroll
        for (int et = 0; et < 4; ++et)
            O[et] = wmma_f16(pa, bv[et], O[et]);

        __syncthreads();   // all reads of buf done before it is re-issued
    }

    // Normalize and write out [B, S, H*E] f32.
    #pragma unroll
    for (int v = 0; v < 8; ++v) {
        const float inv = 1.0f / Lrow[v];
        const int s = qbase + v + (hf << 3);
        float* orow = out + ((size_t)b * SEQ + s) * (N_HEADS * EMBED) + h * EMBED;
        #pragma unroll
        for (int et = 0; et < 4; ++et)
            orow[et * 16 + nn] = O[et][v] * inv;
    }
}

// ---------------------------------------------------------------------------
extern "C" void kernel_launch(void* const* d_in, const int* in_sizes, int n_in,
                              void* d_out, int out_size, void* d_ws, size_t ws_size,
                              hipStream_t stream) {
    const float* x    = (const float*)d_in[0];
    const float* W    = (const float*)d_in[1];
    const float* bias = (const float*)d_in[2];
    float* out = (float*)d_out;

    const size_t head_elems = (size_t)BATCH * N_HEADS * SEQ * EMBED;
    _Float16* qh = (_Float16*)d_ws;
    _Float16* kh = qh + head_elems;
    _Float16* vh = kh + head_elems;

    dim3 g1(QKV_COLS / 64, (BATCH * SEQ) / 64);   // (24, 64)
    qkv_gemm_kernel<<<g1, 128, 0, stream>>>(x, W, bias, qh, kh, vh);

    dim3 g2(BATCH * N_HEADS * (SEQ / 64));        // 512 workgroups
    attn_kernel<<<g2, 128, 0, stream>>>(qh, kh, vh, out);
}
